// heterognn_71090298683528
// MI455X (gfx1250) — matrix operand order repaired and checked by reading, not compiled, not run on hardware
//
#include <hip/hip_runtime.h>
#include <hip/hip_bf16.h>
#include <math.h>

// ---------------------------------------------------------------------------
// Types for CDNA5 WMMA / TDM
// ---------------------------------------------------------------------------
typedef __bf16 bf16_t;
typedef bf16_t bf16x16 __attribute__((ext_vector_type(16)));
typedef bf16_t bf16x8  __attribute__((ext_vector_type(8)));
typedef float  floatx8 __attribute__((ext_vector_type(8)));
typedef unsigned u32x4 __attribute__((ext_vector_type(4)));
typedef unsigned u32x8 __attribute__((ext_vector_type(8)));

#define N_OPENIE 100000
#define N_ENTITY 100000
#define N_EDGE   150000
#define D        512
#define HEADS    8
#define DH       64
#define NEG_SLOPE 0.2f

#define TILE_M 128
#define TILE_N 64
#define TILE_K 32
#define KSTEPS (D / TILE_K)     // 16
#define LDA_F32 36              // padded LDS row stride in f32 (32 data + 4 pad)

// ---------------------------------------------------------------------------
// Tensor Data Mover: async 2D tile load (f32) Global -> LDS.
// 2-group D# (2D tensor): tile = TILE_K x valid_rows, row stride D elems,
// LDS rows padded to LDA_F32 dwords via pad_interval/pad_amount.
// Rows >= valid_rows are zero-filled by TDM OOB handling (M tail).
// ---------------------------------------------------------------------------
__device__ __forceinline__ void tdm_load_a_tile(unsigned lds_addr,
                                                const float* gsrc,
                                                unsigned valid_rows)
{
    const unsigned long long ga = (unsigned long long)(size_t)gsrc;

    u32x4 g0;
    g0[0] = 1u;                                   // count=1 (valid descriptor)
    g0[1] = lds_addr;                             // LDS byte address
    g0[2] = (unsigned)ga;                         // global_addr[31:0]
    g0[3] = (unsigned)((ga >> 32) & 0x01FFFFFFu)  // global_addr[56:32]
          | (2u << 30);                           // type = 2 ("image")

    u32x8 g1;
    g1[0] = (2u << 16)        // data_size = 4 bytes
          | (1u << 20)        // pad_enable
          | (4u << 22)        // pad_interval: every 32 DWORDs
          | (3u << 25);       // pad_amount: 4 DWORDs (16B) -> 36-dword rows
    g1[1] = ((unsigned)TILE_K & 0xFFFFu) << 16;        // tensor_dim0[15:0] = 32
    g1[2] = ((unsigned)TILE_K >> 16)                   // tensor_dim0[31:16]
          | (valid_rows << 16);                        // tensor_dim1[15:0]
    g1[3] = (valid_rows >> 16)                         // tensor_dim1[31:16]
          | ((unsigned)TILE_K << 16);                  // tile_dim0 = 32
    g1[4] = (unsigned)TILE_M;                          // tile_dim1 = 128, tile_dim2 = 0
    g1[5] = (unsigned)D;                               // tensor_dim0_stride = 512
    g1[6] = 0u;
    g1[7] = 0u;

    asm volatile("tensor_load_to_lds %0, %1" :: "s"(g0), "s"(g1) : "memory");
}

// ---------------------------------------------------------------------------
// Pre-transpose + convert W[K=512][N=512] f32 -> Wt[N][K] bf16 (L2-resident).
// ---------------------------------------------------------------------------
__global__ void wtrans_bf16(const float* __restrict__ W, bf16_t* __restrict__ Wt)
{
    int g = blockIdx.x * blockDim.x + threadIdx.x;
    if (g >= D * D) return;
    int k = g >> 9, n = g & (D - 1);
    Wt[(size_t)n * D + k] = (bf16_t)W[g];
}

// ---------------------------------------------------------------------------
// Tiled bf16 WMMA GEMM:  C[M x 512] = A[M x 512] @ W[512 x 512] + bias
//  - A tiles (128 x 32 f32) streamed by the TDM into double-buffered LDS,
//    converted f32->bf16 at fragment-build time.
//  - B fragments read directly from pre-converted bf16 Wt[N][K] in global
//    (512 KB, L2-resident; two aligned 16B loads per fragment).
//  - 256 threads = 8 waves; wave w owns rows [w*16, w*16+16) x 64 N cols
//    => 4 accumulators of v_wmma_f32_16x16x32_bf16 per K-step.
// TANH_REDUCE epilogue accumulates sum over rows of tanh(C + bias) into
// red_acc[512] instead of storing C.
// ---------------------------------------------------------------------------
template <bool TANH_REDUCE>
__global__ __launch_bounds__(256)
void gemm512_bf16wmma(const float* __restrict__ A,
                      const bf16_t* __restrict__ Wt,   // [N=512][K=512] bf16
                      const float* __restrict__ bias,  // [512]
                      float* __restrict__ Cout,        // used if !TANH_REDUCE
                      float* __restrict__ red_acc,     // used if TANH_REDUCE
                      int M)
{
    __shared__ float lA[2][TILE_M * LDA_F32];   // 2 x 18KB, TDM-filled

    const int tid  = threadIdx.x;
    const int lane = tid & 31;
    const int wave = tid >> 5;
    const int m0   = blockIdx.x * TILE_M;
    const int n0   = blockIdx.y * TILE_N;

    const unsigned validRows =
        ((M - m0) < TILE_M) ? (unsigned)(M - m0) : (unsigned)TILE_M;

    floatx8 acc[4];
#pragma unroll
    for (int f = 0; f < 4; ++f)
#pragma unroll
        for (int i = 0; i < 8; ++i) acc[f][i] = 0.0f;

    // CDNA5 wave32 WMMA fragment addressing
    const int mrow = wave * 16 + (lane & 15);   // A row owned by this lane
    const int ka   = (lane >> 4) * 8;           // A: upper half-wave -> K+8
    const int kb2  = (lane >> 4) * 8;           // B: upper half-wave -> K+8

    // prologue: kick off first A tile
    if (wave == 0)
        tdm_load_a_tile((unsigned)(size_t)&lA[0][0], A + (size_t)m0 * D, validRows);

    for (int step = 0; step < KSTEPS; ++step) {
        const int cur = step & 1;
        if (wave == 0) {
            if (step + 1 < KSTEPS) {
                tdm_load_a_tile((unsigned)(size_t)&lA[cur ^ 1][0],
                                A + (size_t)m0 * D + (step + 1) * TILE_K, validRows);
                __builtin_amdgcn_s_wait_tensorcnt(1);  // current tile complete
            } else {
                __builtin_amdgcn_s_wait_tensorcnt(0);
            }
        }
        __syncthreads();   // current buffer visible to all waves

        // ---- A fragment: 4 aligned LDS b128 loads + pack to bf16 ----------
        const float* arow = &lA[cur][mrow * LDA_F32];
        bf16x16 afrag;
        {
            const float4 a0 = *(const float4*)(arow + ka);
            const float4 a1 = *(const float4*)(arow + ka + 4);
            const float4 a2 = *(const float4*)(arow + ka + 16);
            const float4 a3 = *(const float4*)(arow + ka + 20);
            afrag[0]  = (bf16_t)a0.x; afrag[1]  = (bf16_t)a0.y;
            afrag[2]  = (bf16_t)a0.z; afrag[3]  = (bf16_t)a0.w;
            afrag[4]  = (bf16_t)a1.x; afrag[5]  = (bf16_t)a1.y;
            afrag[6]  = (bf16_t)a1.z; afrag[7]  = (bf16_t)a1.w;
            afrag[8]  = (bf16_t)a2.x; afrag[9]  = (bf16_t)a2.y;
            afrag[10] = (bf16_t)a2.z; afrag[11] = (bf16_t)a2.w;
            afrag[12] = (bf16_t)a3.x; afrag[13] = (bf16_t)a3.y;
            afrag[14] = (bf16_t)a3.z; afrag[15] = (bf16_t)a3.w;
        }

        // ---- 4 WMMAs over the 64-wide N strip; B straight from L2 ---------
        const int k0 = step * TILE_K;
#pragma unroll
        for (int f = 0; f < 4; ++f) {
            const bf16x8* wp = (const bf16x8*)(Wt +
                (size_t)(n0 + f * 16 + (lane & 15)) * D + k0 + kb2);
            bf16x8 blo = wp[0];
            bf16x8 bhi = wp[2];
            bf16x16 bfrag;
#pragma unroll
            for (int i = 0; i < 8; ++i) { bfrag[i] = blo[i]; bfrag[8 + i] = bhi[i]; }
            acc[f] = __builtin_amdgcn_wmma_f32_16x16x32_bf16(
                false, afrag, false, bfrag, (short)0, acc[f], false, false);
        }
        __syncthreads();   // all reads of 'cur' done before TDM refills it
    }

    // ---- epilogue ---------------------------------------------------------
    if (!TANH_REDUCE) {
#pragma unroll
        for (int f = 0; f < 4; ++f) {
            const int col = n0 + f * 16 + (lane & 15);
            const float bcol = bias[col];
#pragma unroll
            for (int r = 0; r < 8; ++r) {
                const int grow = m0 + wave * 16 + r + ((lane >> 4) << 3);
                if (grow < M)
                    Cout[(size_t)grow * D + col] = acc[f][r] + bcol;
            }
        }
    } else {
#pragma unroll
        for (int f = 0; f < 4; ++f) {
            const int col = n0 + f * 16 + (lane & 15);
            const float bcol = bias[col];
            float s = 0.0f;
#pragma unroll
            for (int r = 0; r < 8; ++r) {
                const int grow = m0 + wave * 16 + r + ((lane >> 4) << 3);
                if (grow < M) s += tanhf(acc[f][r] + bcol);
            }
            s += __shfl_xor(s, 16, 32);   // combine the two half-wave row groups
            if (lane < 16) atomicAdd(&red_acc[col], s);
        }
    }
}

// ---------------------------------------------------------------------------
// fills
// ---------------------------------------------------------------------------
__global__ void fill_f32(float* p, float v, size_t n) {
    size_t i = (size_t)blockIdx.x * blockDim.x + threadIdx.x;
    if (i < n) p[i] = v;
}
__global__ void fill_u32(unsigned* p, unsigned v, size_t n) {
    size_t i = (size_t)blockIdx.x * blockDim.x + threadIdx.x;
    if (i < n) p[i] = v;
}

// ---------------------------------------------------------------------------
// Per-(node, head) attention logits: out[n,h] = dot(h[n,h,:], att[h,:])
// ---------------------------------------------------------------------------
__global__ void logits_kernel(const float* __restrict__ h, int n_nodes,
                              const float* __restrict__ a0, float* __restrict__ o0,
                              const float* __restrict__ a1, float* __restrict__ o1,
                              const float* __restrict__ a2, float* __restrict__ o2)
{
    int g = blockIdx.x * blockDim.x + threadIdx.x;
    if (g >= n_nodes * HEADS) return;
    int n = g >> 3, hh = g & 7;
    const float* hp = h + (size_t)n * D + hh * DH;
    float s0 = 0.f, s1 = 0.f, s2 = 0.f;
#pragma unroll 8
    for (int d = 0; d < DH; ++d) {
        float v = hp[d];
        s0 += v * a0[hh * DH + d];
        if (a1) s1 += v * a1[hh * DH + d];
        if (a2) s2 += v * a2[hh * DH + d];
    }
    o0[g] = s0;
    if (o1) o1[g] = s1;
    if (o2) o2[g] = s2;
}

// order-preserving float <-> uint key for integer atomicMax-based segment max
__device__ __forceinline__ unsigned f2key(float f) {
    unsigned u = __float_as_uint(f);
    return (u & 0x80000000u) ? ~u : (u | 0x80000000u);
}
__device__ __forceinline__ float key2f(unsigned k) {
    unsigned u = (k & 0x80000000u) ? (k & 0x7FFFFFFFu) : ~k;
    return __uint_as_float(u);
}

// pass 1: alpha = leaky_relu(a_src[src] + a_dst[dst]); segment max via atomicMax
__global__ void edge_alpha_max(const int* __restrict__ edges, int E,
                               const float* __restrict__ asrc,
                               const float* __restrict__ adst,
                               float* __restrict__ alpha,
                               unsigned* __restrict__ amax_key)
{
    int g = blockIdx.x * blockDim.x + threadIdx.x;
    if (g >= E * HEADS) return;
    int e = g >> 3, hh = g & 7;
    int s = edges[e], d = edges[E + e];
    float a = asrc[s * HEADS + hh] + adst[d * HEADS + hh];
    a = (a > 0.f) ? a : NEG_SLOPE * a;
    alpha[g] = a;
    atomicMax(&amax_key[d * HEADS + hh], f2key(a));
}

// pass 2: ex = exp(alpha - amax[dst]); denom += ex  (ex overwrites alpha)
__global__ void edge_exp_denom(const int* __restrict__ edges, int E,
                               float* __restrict__ alpha,
                               const unsigned* __restrict__ amax_key,
                               float* __restrict__ denom)
{
    int g = blockIdx.x * blockDim.x + threadIdx.x;
    if (g >= E * HEADS) return;
    int e = g >> 3, hh = g & 7;
    int d = edges[E + e];
    float amax = key2f(amax_key[d * HEADS + hh]);
    float ex = expf(alpha[g] - amax);
    alpha[g] = ex;
    atomicAdd(&denom[d * HEADS + hh], ex);
}

// pass 3: out[dst] += h_src[src] * (ex / (denom[dst] + 1e-16)).
__global__ void edge_scatter(const int* __restrict__ edges, int E,
                             const float* __restrict__ hsrc,
                             const float* __restrict__ alpha,   // holds ex
                             const float* __restrict__ denom,
                             float* __restrict__ out)
{
    int g = blockIdx.x * blockDim.x + threadIdx.x;
    int e = g >> 7;
    if (e >= E) return;
    int t  = g & 127;
    int hh = t >> 4;
    int db = (t & 15) * 4;
    int s = edges[e], d = edges[E + e];
    float w = alpha[e * HEADS + hh] / (denom[d * HEADS + hh] + 1e-16f);
    const float4 hv = *(const float4*)(hsrc + (size_t)s * D + hh * DH + db);
    float* op = out + (size_t)d * D + hh * DH + db;
    atomicAdd(op + 0, hv.x * w);
    atomicAdd(op + 1, hv.y * w);
    atomicAdd(op + 2, hv.z * w);
    atomicAdd(op + 3, hv.w * w);
}

// relu in place + column sums (global mean-pool numerator). 32 rows / block.
__global__ void relu_colsum(float* __restrict__ buf, int n_nodes,
                            float* __restrict__ colsum)
{
    int r0 = blockIdx.x * 32;
    int t  = threadIdx.x;   // 256 threads; cols t and t+256
    float s0 = 0.f, s1 = 0.f;
    for (int r = 0; r < 32; ++r) {
        int row = r0 + r;
        if (row >= n_nodes) break;
        size_t base = (size_t)row * D;
        float v0 = buf[base + t];       v0 = v0 > 0.f ? v0 : 0.f;
        float v1 = buf[base + 256 + t]; v1 = v1 > 0.f ? v1 : 0.f;
        buf[base + t] = v0;  buf[base + 256 + t] = v1;
        s0 += v0; s1 += v1;
    }
    atomicAdd(&colsum[t], s0);
    atomicAdd(&colsum[256 + t], s1);
}

__device__ __forceinline__ float block_reduce_512(float v, float* red, int t) {
    red[t] = v; __syncthreads();
#pragma unroll
    for (int off = 256; off > 0; off >>= 1) {
        if (t < off) red[t] += red[t + off];
        __syncthreads();
    }
    float r = red[0];
    __syncthreads();
    return r;
}

// semantic softmax + fused pooling + Linear(512, 2). One block, 512 threads.
__global__ void final_head(const float* __restrict__ kacc,   // [2][512]
                           const float* __restrict__ q,      // q_sem [512]
                           const float* __restrict__ cs0,    // colsum out0
                           const float* __restrict__ cs1,    // colsum out1
                           const float* __restrict__ linw,   // [512][2]
                           const float* __restrict__ linb,   // [2]
                           float* __restrict__ out, float invN)
{
    __shared__ float red[512];
    int t = threadIdx.x;
    float s0 = block_reduce_512(kacc[t]       * invN * q[t], red, t);
    float s1 = block_reduce_512(kacc[512 + t] * invN * q[t], red, t);
    float m  = fmaxf(s0, s1);
    float e0 = expf(s0 - m), e1 = expf(s1 - m);
    float sem0 = e0 / (e0 + e1), sem1 = e1 / (e0 + e1);
    float pooled = (sem0 * cs0[t] + sem1 * cs1[t]) * invN;
    float o0 = block_reduce_512(pooled * linw[t * 2 + 0], red, t);
    float o1 = block_reduce_512(pooled * linw[t * 2 + 1], red, t);
    if (t == 0) { out[0] = o0 + linb[0]; out[1] = o1 + linb[1]; }
}

// ---------------------------------------------------------------------------
// Orchestration
// ---------------------------------------------------------------------------
extern "C" void kernel_launch(void* const* d_in, const int* in_sizes, int n_in,
                              void* d_out, int out_size, void* d_ws, size_t ws_size,
                              hipStream_t stream)
{
    (void)in_sizes; (void)n_in; (void)out_size; (void)ws_size;

    const float* x_openie = (const float*)d_in[0];
    const float* x_entity = (const float*)d_in[1];
    const int*   edge_e2o = (const int*)d_in[2];
    const int*   edge_o2o = (const int*)d_in[3];
    const float* proj_o_w = (const float*)d_in[4];
    const float* proj_o_b = (const float*)d_in[5];
    const float* proj_e_w = (const float*)d_in[6];
    const float* proj_e_b = (const float*)d_in[7];
    const float* att_src_e2o = (const float*)d_in[8];
    const float* att_dst_e2o = (const float*)d_in[9];
    const float* att_src_o2o = (const float*)d_in[10];
    const float* att_dst_o2o = (const float*)d_in[11];
    const float* k_lin_w = (const float*)d_in[12];
    const float* k_lin_b = (const float*)d_in[13];
    const float* q_sem   = (const float*)d_in[14];
    const float* lin_w   = (const float*)d_in[15];
    const float* lin_b   = (const float*)d_in[16];
    float* outp = (float*)d_out;

    // workspace layout (bytes)
    const size_t SZ_H = (size_t)N_OPENIE * D * sizeof(float);   // 204.8 MB
    char* ws = (char*)d_ws;
    float* hO   = (float*)(ws + 0 * SZ_H);        // h_openie
    float* hE   = (float*)(ws + 1 * SZ_H);        // h_entity, later out1
    float* out0 = (float*)(ws + 2 * SZ_H);
    float* out1 = hE;                             // alias: h_entity dead by then
    char* sm = ws + 3 * SZ_H;
    const size_t SZ_NH = (size_t)N_OPENIE * HEADS * sizeof(float);  // 3.2 MB
    float*    aSrcE2O = (float*)(sm); sm += SZ_NH;
    float*    aDstE2O = (float*)(sm); sm += SZ_NH;
    float*    aSrcO2O = (float*)(sm); sm += SZ_NH;
    float*    aDstO2O = (float*)(sm); sm += SZ_NH;
    unsigned* amaxKey = (unsigned*)(sm); sm += SZ_NH;
    float*    denom   = (float*)(sm); sm += SZ_NH;
    float*    alphaB  = (float*)(sm); sm += (size_t)N_EDGE * HEADS * sizeof(float);
    float*    colsum0 = (float*)(sm); sm += D * sizeof(float);
    float*    colsum1 = (float*)(sm); sm += D * sizeof(float);
    float*    kacc    = (float*)(sm); sm += 2 * D * sizeof(float);
    bf16_t*   WtO     = (bf16_t*)(sm); sm += (size_t)D * D * sizeof(bf16_t);
    bf16_t*   WtE     = (bf16_t*)(sm); sm += (size_t)D * D * sizeof(bf16_t);
    bf16_t*   WtK     = (bf16_t*)(sm); sm += (size_t)D * D * sizeof(bf16_t);

    const dim3 gemmGrid((N_OPENIE + TILE_M - 1) / TILE_M, D / TILE_N);
    const int nhElems = N_OPENIE * HEADS;
    const int ehElems = N_EDGE * HEADS;
    const int wElems  = D * D;

    // 0) weight transpose + bf16 conversion (L2-resident operands for WMMA)
    wtrans_bf16<<<(wElems + 255) / 256, 256, 0, stream>>>(proj_o_w, WtO);
    wtrans_bf16<<<(wElems + 255) / 256, 256, 0, stream>>>(proj_e_w, WtE);
    wtrans_bf16<<<(wElems + 255) / 256, 256, 0, stream>>>(k_lin_w, WtK);

    // 1) per-type projections (TDM-fed bf16 WMMA GEMMs)
    gemm512_bf16wmma<false><<<gemmGrid, 256, 0, stream>>>(
        x_openie, WtO, proj_o_b, hO, nullptr, N_OPENIE);
    gemm512_bf16wmma<false><<<gemmGrid, 256, 0, stream>>>(
        x_entity, WtE, proj_e_b, hE, nullptr, N_ENTITY);

    // 2) attention logits
    logits_kernel<<<(nhElems + 255) / 256, 256, 0, stream>>>(
        hE, N_ENTITY, att_src_e2o, aSrcE2O, nullptr, nullptr, nullptr, nullptr);
    logits_kernel<<<(nhElems + 255) / 256, 256, 0, stream>>>(
        hO, N_OPENIE, att_dst_e2o, aDstE2O, att_src_o2o, aSrcO2O, att_dst_o2o, aDstO2O);

    // 3) metapath 0: entity -> openie
    fill_u32<<<(nhElems + 255) / 256, 256, 0, stream>>>(amaxKey, 0u, nhElems);
    fill_f32<<<(nhElems + 255) / 256, 256, 0, stream>>>(denom, 0.f, nhElems);
    fill_f32<<<((size_t)N_OPENIE * D + 255) / 256, 256, 0, stream>>>(
        out0, 0.f, (size_t)N_OPENIE * D);
    edge_alpha_max<<<(ehElems + 255) / 256, 256, 0, stream>>>(
        edge_e2o, N_EDGE, aSrcE2O, aDstE2O, alphaB, amaxKey);
    edge_exp_denom<<<(ehElems + 255) / 256, 256, 0, stream>>>(
        edge_e2o, N_EDGE, alphaB, amaxKey, denom);
    edge_scatter<<<((size_t)N_EDGE * 128 + 255) / 256, 256, 0, stream>>>(
        edge_e2o, N_EDGE, hE, alphaB, denom, out0);

    // 4) metapath 1: openie -> openie (out1 reuses hE's buffer; hE is dead)
    fill_u32<<<(nhElems + 255) / 256, 256, 0, stream>>>(amaxKey, 0u, nhElems);
    fill_f32<<<(nhElems + 255) / 256, 256, 0, stream>>>(denom, 0.f, nhElems);
    fill_f32<<<((size_t)N_OPENIE * D + 255) / 256, 256, 0, stream>>>(
        out1, 0.f, (size_t)N_OPENIE * D);
    edge_alpha_max<<<(ehElems + 255) / 256, 256, 0, stream>>>(
        edge_o2o, N_EDGE, aSrcO2O, aDstO2O, alphaB, amaxKey);
    edge_exp_denom<<<(ehElems + 255) / 256, 256, 0, stream>>>(
        edge_o2o, N_EDGE, alphaB, amaxKey, denom);
    edge_scatter<<<((size_t)N_EDGE * 128 + 255) / 256, 256, 0, stream>>>(
        edge_o2o, N_EDGE, hO, alphaB, denom, out1);

    // 5) relu in place + column sums (mean-pool numerators)
    fill_f32<<<(2 * D + 255) / 256, 256, 0, stream>>>(colsum0, 0.f, D);
    fill_f32<<<(2 * D + 255) / 256, 256, 0, stream>>>(colsum1, 0.f, D);
    fill_f32<<<(2 * D + 255) / 256, 256, 0, stream>>>(kacc, 0.f, 2 * D);
    relu_colsum<<<(N_OPENIE + 31) / 32, 256, 0, stream>>>(out0, N_OPENIE, colsum0);
    relu_colsum<<<(N_OPENIE + 31) / 32, 256, 0, stream>>>(out1, N_OPENIE, colsum1);

    // 6) semantic attention keys: kacc[m] = sum_n tanh(out_m @ k_lin_w + b)
    gemm512_bf16wmma<true><<<gemmGrid, 256, 0, stream>>>(
        out0, WtK, k_lin_b, nullptr, kacc, N_OPENIE);
    gemm512_bf16wmma<true><<<gemmGrid, 256, 0, stream>>>(
        out1, WtK, k_lin_b, nullptr, kacc + D, N_OPENIE);

    // 7) softmax over metapaths + fused pooled + Linear(512, 2)
    final_head<<<1, 512, 0, stream>>>(kacc, q_sem, colsum0, colsum1,
                                      lin_w, lin_b, outp, 1.0f / (float)N_OPENIE);
}
